// PruneGAT_34041910788165
// MI455X (gfx1250) — matrix operand order repaired and checked by reading, not compiled, not run on hardware
//
#include <hip/hip_runtime.h>
#include <hip/hip_bf16.h>
#include <math.h>

// ---------------------------------------------------------------------------
// PruneGAT on MI455X (gfx1250, wave32).
//   h = X@W on v_wmma_f32_16x16x32_bf16, A tile staged to LDS with the
//   Tensor Data Mover (tensor_load_to_lds + s_wait_tensorcnt), B from
//   pre-transposed bf16 W via b128 loads.
//   Edge pruning: per-src argmax, first-max-wins via packed u64 atomicMax.
//   out = elu(h[best_dst] * drop_mask)   (softmax over singleton axis == 1,
//   so the attention vector `a` is numerically dead).
// ---------------------------------------------------------------------------

#define NN   10000
#define EE   320000
#define KIN  512
#define KOUT 256

typedef __attribute__((ext_vector_type(16))) __bf16       v16bf;
typedef __attribute__((ext_vector_type(8)))  __bf16       v8bf;
typedef __attribute__((ext_vector_type(8)))  float        v8f;
typedef __attribute__((ext_vector_type(4)))  unsigned int u32x4;
typedef __attribute__((ext_vector_type(8)))  unsigned int u32x8;

// f32 -> bf16, round-to-nearest-even (bit exact, no __bf16 arithmetic needed)
__device__ __forceinline__ __bf16 f2bf(float f) {
    unsigned int u = __float_as_uint(f);
    unsigned int r = (u + 0x7fffu + ((u >> 16) & 1u)) >> 16;
    return __builtin_bit_cast(__bf16, (unsigned short)r);
}

// byte offset of a __shared__ object within the workgroup LDS allocation
__device__ __forceinline__ unsigned lds_byte_offset(const void* p) {
    return (unsigned)(unsigned long long)
        (__attribute__((address_space(3))) const char*)p;
}

// --------------------------- edge argmax ------------------------------------
__global__ void prunegat_init_keys(unsigned long long* __restrict__ keys) {
    int i = blockIdx.x * blockDim.x + threadIdx.x;
    if (i < NN) keys[i] = 0ull;
}

__global__ void prunegat_edge_scan(const float* __restrict__ edges_prob,
                                   const int* __restrict__ edge_index,
                                   unsigned long long* __restrict__ keys) {
    int e = blockIdx.x * blockDim.x + threadIdx.x;
    if (e >= EE) return;
    int s = edge_index[e];
    int d = edge_index[EE + e];
    float p = edges_prob[(size_t)s * NN + d];           // uniform [0,1): bits monotone
    unsigned long long key =
        ((unsigned long long)__float_as_uint(p) << 32) |
        (unsigned long long)(unsigned int)(EE - e);     // larger low word = earlier edge
    atomicMax(keys + s, key);
}

__global__ void prunegat_extract(const unsigned long long* __restrict__ keys,
                                 const int* __restrict__ edge_index,
                                 int* __restrict__ best_dst) {
    int i = blockIdx.x * blockDim.x + threadIdx.x;
    if (i >= NN) return;
    int e = EE - (int)(unsigned int)(keys[i] & 0xffffffffull);
    e = min(max(e, 0), EE - 1);                         // every node has >=1 edge
    best_dst[i] = edge_index[EE + e];
}

// --------------------------- bf16 pre-passes --------------------------------
__global__ void prunegat_conv_x(const float4* __restrict__ X4,
                                __bf16* __restrict__ Xbf) {
    int i = blockIdx.x * blockDim.x + threadIdx.x;
    if (i >= (NN * KIN) / 4) return;
    float4 v = X4[i];
    __bf16* p = Xbf + (size_t)i * 4;
    p[0] = f2bf(v.x); p[1] = f2bf(v.y); p[2] = f2bf(v.z); p[3] = f2bf(v.w);
}

// W[k][n] (f32) -> Wt[n][k] (bf16): per-output-column contiguous in K
__global__ void prunegat_conv_wt(const float* __restrict__ W,
                                 __bf16* __restrict__ Wt) {
    int i = blockIdx.x * blockDim.x + threadIdx.x;
    if (i >= KIN * KOUT) return;
    int k = i / KOUT, n = i % KOUT;                     // coalesced read of W
    Wt[(size_t)n * KIN + k] = f2bf(W[i]);
}

// --------------------------- GEMM: h = X @ W (WMMA bf16) --------------------
// 16 waves/block; block owns rows [16*bx, 16*bx+16), wave w owns cols [16w,16w+16).
// A tile (16 x 512 bf16 = 16 KB) is DMA'd to LDS once per block by the TDM.
__global__ __launch_bounds__(512) void prunegat_gemm(const __bf16* __restrict__ Xbf,
                                                     const __bf16* __restrict__ Wt,
                                                     float* __restrict__ h) {
    __shared__ __align__(128) __bf16 atile[16 * KIN];   // 16 KB

    const int wave = threadIdx.x >> 5;
    const int lane = threadIdx.x & 31;
    const int l16  = lane & 15;
    const int half = lane >> 4;                         // 0 or 1
    const int m0   = blockIdx.x * 16;
    const int n0   = wave * 16;

    // ---- TDM: one descriptor, issued by wave 0 only (EXEC-independent DMA) ----
    if (wave == 0) {
        unsigned long long ga =
            (unsigned long long)Xbf + (unsigned long long)m0 * (KIN * 2);
        u32x4 g0;
        g0[0] = 1u;                                     // count=1, user descriptor
        g0[1] = lds_byte_offset(atile);                 // lds_addr
        g0[2] = (unsigned)ga;                           // global_addr[31:0]
        g0[3] = ((unsigned)(ga >> 32) & 0x01ffffffu)    // global_addr[56:32]
              | 0x80000000u;                            // type=2 ("image")
        u32x8 g1;
        g1[0] = 0x00010000u;                            // wg_mask=0, data_size=1 (2B)
        g1[1] = (KIN & 0xffffu) << 16;                  // tensor_dim0[15:0]=512
        g1[2] = (NN  & 0xffffu) << 16;                  // dim0 hi=0 | tensor_dim1 lo=10000
        g1[3] = ((unsigned)KIN) << 16;                  // dim1 hi=0 | tile_dim0=512
        g1[4] = 16u;                                    // tile_dim1=16, tile_dim2=0
        g1[5] = (unsigned)KIN;                          // tensor_dim0_stride=512
        g1[6] = 0u;
        g1[7] = 0u;
        asm volatile("tensor_load_to_lds %0, %1" :: "s"(g0), "s"(g1) : "memory");
#if __has_builtin(__builtin_amdgcn_s_wait_tensorcnt)
        __builtin_amdgcn_s_wait_tensorcnt(0);
#else
        asm volatile("s_wait_tensorcnt 0x0" ::: "memory");
#endif
    }
    __syncthreads();

    // lane-private operand bases; A and B share the contiguous K order k = 16*half + e
    const __bf16* arow = atile + (size_t)l16 * KIN;      // A row l16
    const int     ka   = half * 16;                      // A K sub-offset
    const __bf16* wcol = Wt + (size_t)(n0 + l16) * KIN + half * 16;  // B col, K base

    v8f acc = {};
#pragma unroll
    for (int k0 = 0; k0 < KIN; k0 += 32) {
        // A 16x32 bf16: elem e -> K = k0 + 16*half + e (contiguous)
        v8bf alo = *(const v8bf*)(arow + k0 + ka);        // ds_load_b128
        v8bf ahi = *(const v8bf*)(arow + k0 + ka + 8);    // ds_load_b128
        v16bf av = __builtin_shufflevector(alo, ahi,
                       0, 1, 2, 3, 4, 5, 6, 7, 8, 9, 10, 11, 12, 13, 14, 15);
        // B 32x16 bf16: elem e -> K = k0 + 16*half + e (contiguous in Wt)
        v16bf bv = *(const v16bf*)(wcol + k0);            // 2x global_load_b128

        acc = __builtin_amdgcn_wmma_f32_16x16x32_bf16(
            /*neg_a=*/false, av, /*neg_b=*/false, bv,
            /*c_mod=*/(short)0, acc, /*reuse_a=*/false, /*reuse_b=*/false);
    }

    // D 16x16 f32: VGPR r -> row m0 + 8*half + r, column n0+l16
    float* hp = h + (size_t)(m0 + 8 * half) * KOUT + n0 + l16;
#pragma unroll
    for (int r = 0; r < 8; ++r) hp[(size_t)r * KOUT] = acc[r];
}

// --------------------------- epilogue ---------------------------------------
__global__ void prunegat_finish(const float* __restrict__ h,
                                const int* __restrict__ best_dst,
                                const int* __restrict__ cls_idx,
                                const float* __restrict__ drop_mask,
                                float* __restrict__ out) {
    int idx = blockIdx.x * blockDim.x + threadIdx.x;
    if (idx >= NN * KOUT) return;
    int c = idx >> 8;                                   // KOUT == 256
    int j = idx & 255;
    int rowj = best_dst[cls_idx[c]];
    float v = h[(size_t)rowj * KOUT + j] * drop_mask[idx];
    out[idx] = v > 0.0f ? v : expm1f(v);                // ELU(alpha=1)
}

// ---------------------------------------------------------------------------
extern "C" void kernel_launch(void* const* d_in, const int* in_sizes, int n_in,
                              void* d_out, int out_size, void* d_ws, size_t ws_size,
                              hipStream_t stream) {
    const float* X          = (const float*)d_in[0];    // [N, IN]
    const float* W          = (const float*)d_in[1];    // [IN, OUT]
    /* d_in[2] = a : numerically dead (softmax over singleton axis) */
    const float* edges_prob = (const float*)d_in[3];    // [N, N]
    const float* drop_mask  = (const float*)d_in[4];    // [N, OUT]
    const int*   edge_index = (const int*)d_in[5];      // [2, E]
    const int*   cls_idx    = (const int*)d_in[6];      // [N]
    float*       out        = (float*)d_out;            // [N, OUT]

    // workspace carve-up (128B aligned sections), total ~20.9 MB
    char* ws = (char*)d_ws;
    unsigned long long* keys = (unsigned long long*)(ws);          // 80000 B
    int*    best_dst         = (int*)(ws + 80000);                 // 40000 B
    __bf16* Wt               = (__bf16*)(ws + 120064);             // 262144 B
    __bf16* Xbf              = (__bf16*)(ws + 382208);             // 10240000 B
    float*  h                = (float*)(ws + 10622208);            // 10240000 B

    prunegat_conv_x   <<<(NN * KIN / 4 + 255) / 256, 256, 0, stream>>>(
                        (const float4*)X, Xbf);
    prunegat_conv_wt  <<<(KIN * KOUT + 255) / 256, 256, 0, stream>>>(W, Wt);
    prunegat_init_keys<<<(NN + 255) / 256, 256, 0, stream>>>(keys);
    prunegat_edge_scan<<<(EE + 255) / 256, 256, 0, stream>>>(edges_prob, edge_index, keys);
    prunegat_extract  <<<(NN + 255) / 256, 256, 0, stream>>>(keys, edge_index, best_dst);
    prunegat_gemm     <<<NN / 16, 512, 0, stream>>>(Xbf, Wt, h);
    prunegat_finish   <<<(NN * KOUT + 255) / 256, 256, 0, stream>>>(h, best_dst, cls_idx,
                                                                    drop_mask, out);
}
